// Blipv2_31782757990438
// MI455X (gfx1250) — matrix operand order repaired and checked by reading, not compiled, not run on hardware
//
#include <hip/hip_runtime.h>
#include <hip/hip_bf16.h>

typedef __attribute__((ext_vector_type(16))) _Float16 v16h;
typedef __attribute__((ext_vector_type(8)))  float    v8f;

#define BB 8
#define TT 512
#define ZZ 16
#define II 1024
#define DD 256
#define LDSPITCH 264   // 256 halves + 8 pad -> 132 dwords/row (4 mod 64 banks)

union HFrag { v16h v; uint4 q[2]; };

// ---------------------------------------------------------------------------
// WMMA similarity + fused argmax over I, with LDS-staged f16 image tiles.
// grid = B*Z*(T/128) blocks, 256 threads = 8 waves; wave w owns text rows
// [t0, t0+16). Each 16x256 image tile is converted f32->f16 once per block
// into LDS (row norms computed from the f32 data on the way in), then all 8
// waves read B fragments from LDS.
// ---------------------------------------------------------------------------
__global__ __launch_bounds__(256) void argmax_wmma_kernel(
    const float* __restrict__ text,    // [B,T,D]
    const float* __restrict__ image,   // [B,Z,I,D]
    int* __restrict__ idx_out)         // [B,T,Z]
{
    __shared__ _Float16 tile[2][16][LDSPITCH];  // 2 x 8448 B
    __shared__ float    snorm[2][16];           // 1/max(||row||, eps)

    const int tid    = (int)threadIdx.x;
    const int lane   = tid & 31;
    const int wave   = tid >> 5;
    const int tchunk = (int)(blockIdx.x & 3);        // T/128 = 4 chunks
    const int bz     = (int)(blockIdx.x >> 2);       // [0, B*Z)
    const int b      = bz >> 4;                      // Z = 16
    const int z      = bz & 15;
    const int t0     = tchunk * 128 + wave * 16;
    const int nloc   = lane & 15;                    // row (A) / col (B) in tile
    const int khalf  = lane >> 4;

    // staging role: thread handles row srow, 16-float segment skseg
    const int srow  = tid >> 4;                      // 0..15
    const int skseg = (tid & 15) << 4;               // 0,16,...,240

    // ---- Load A fragments: text rows t0..t0+15, K = 0..255 --------------
    // A 16x32 f16 layout: lane m = lane%16; lanes 0-15 hold K {0..7,16..23},
    // lanes 16-31 hold K {8..15,24..31}  -> two 8-float chunks per lane.
    const float* trow = text + (size_t)(b * TT + t0 + nloc) * DD;
    v16h afrag[8];
#pragma unroll
    for (int s = 0; s < 8; ++s) {
        const int kb = s * 32 + khalf * 8;
        const float4 c0 = *reinterpret_cast<const float4*>(trow + kb);
        const float4 c1 = *reinterpret_cast<const float4*>(trow + kb + 4);
        const float4 c2 = *reinterpret_cast<const float4*>(trow + kb + 16);
        const float4 c3 = *reinterpret_cast<const float4*>(trow + kb + 20);
        afrag[s][0]  = (_Float16)c0.x; afrag[s][1]  = (_Float16)c0.y;
        afrag[s][2]  = (_Float16)c0.z; afrag[s][3]  = (_Float16)c0.w;
        afrag[s][4]  = (_Float16)c1.x; afrag[s][5]  = (_Float16)c1.y;
        afrag[s][6]  = (_Float16)c1.z; afrag[s][7]  = (_Float16)c1.w;
        afrag[s][8]  = (_Float16)c2.x; afrag[s][9]  = (_Float16)c2.y;
        afrag[s][10] = (_Float16)c2.z; afrag[s][11] = (_Float16)c2.w;
        afrag[s][12] = (_Float16)c3.x; afrag[s][13] = (_Float16)c3.y;
        afrag[s][14] = (_Float16)c3.z; afrag[s][15] = (_Float16)c3.w;
    }

    const float* im_bz = image + (size_t)(b * ZZ + z) * II * DD;

    // cooperative stage of tile i0 into LDS buffer `buf`
    auto stage = [&](int i0, int buf) {
        const float* g = im_bz + (size_t)(i0 + srow) * DD + skseg;
        const float4 f0 = *reinterpret_cast<const float4*>(g);
        const float4 f1 = *reinterpret_cast<const float4*>(g + 4);
        const float4 f2 = *reinterpret_cast<const float4*>(g + 8);
        const float4 f3 = *reinterpret_cast<const float4*>(g + 12);
        // row sum of squares from f32 data (threads sharing a row are one
        // half-wave: lanes 0-15 / 16-31)
        float ss = f0.x*f0.x + f0.y*f0.y + f0.z*f0.z + f0.w*f0.w
                 + f1.x*f1.x + f1.y*f1.y + f1.z*f1.z + f1.w*f1.w
                 + f2.x*f2.x + f2.y*f2.y + f2.z*f2.z + f2.w*f2.w
                 + f3.x*f3.x + f3.y*f3.y + f3.z*f3.z + f3.w*f3.w;
#pragma unroll
        for (int off = 8; off > 0; off >>= 1)
            ss += __shfl_xor(ss, off, 32);
        if ((tid & 15) == 0)
            snorm[buf][srow] = 1.0f / fmaxf(sqrtf(ss), 1e-12f);
        HFrag u;
        u.v[0]  = (_Float16)f0.x; u.v[1]  = (_Float16)f0.y;
        u.v[2]  = (_Float16)f0.z; u.v[3]  = (_Float16)f0.w;
        u.v[4]  = (_Float16)f1.x; u.v[5]  = (_Float16)f1.y;
        u.v[6]  = (_Float16)f1.z; u.v[7]  = (_Float16)f1.w;
        u.v[8]  = (_Float16)f2.x; u.v[9]  = (_Float16)f2.y;
        u.v[10] = (_Float16)f2.z; u.v[11] = (_Float16)f2.w;
        u.v[12] = (_Float16)f3.x; u.v[13] = (_Float16)f3.y;
        u.v[14] = (_Float16)f3.z; u.v[15] = (_Float16)f3.w;
        *reinterpret_cast<uint4*>(&tile[buf][srow][skseg])     = u.q[0];
        *reinterpret_cast<uint4*>(&tile[buf][srow][skseg + 8]) = u.q[1];
    };

    float best[8];
    int   bidx[8];
#pragma unroll
    for (int r = 0; r < 8; ++r) { best[r] = -3.4e38f; bidx[r] = 0; }

    stage(0, 0);   // prologue

    // ---- Sweep I = 1024 image tokens, 64 tiles, double-buffered ---------
    for (int it = 0; it < 64; ++it) {
        const int cur = it & 1;
        __syncthreads();                    // publish tile/snorm[cur]
        if (it < 63) stage((it + 1) * 16, cur ^ 1);

        // B 32x16 f16 layout: lane n = lane%16; lanes 0-15 hold K {0..7,16..23},
        // lanes 16-31 hold K {8..15,24..31} -> two 16-byte LDS chunks per lane.
        v8f acc = {};
#pragma unroll
        for (int s = 0; s < 8; ++s) {
            const int kb = s * 32 + khalf * 8;
            HFrag u;
            u.q[0] = *reinterpret_cast<const uint4*>(&tile[cur][nloc][kb]);
            u.q[1] = *reinterpret_cast<const uint4*>(&tile[cur][nloc][kb + 16]);
            acc = __builtin_amdgcn_wmma_f32_16x16x32_f16(
                /*neg_a=*/false, afrag[s], /*neg_b=*/false, u.v,
                /*c_mod=*/(short)0, acc, /*reuse_a=*/false, /*reuse_b=*/false);
        }
        // cosine scale for this lane's column, then running max (strict '>'
        // keeps first occurrence, matching jnp.argmax over ascending i)
        const float scale = snorm[cur][nloc];
        const int   icol  = it * 16 + nloc;
#pragma unroll
        for (int r = 0; r < 8; ++r) {
            const float v = acc[r] * scale;
            if (v > best[r]) { best[r] = v; bidx[r] = icol; }
        }
    }

    // ---- Reduce across the 16 lanes of each half-wave (N dimension) -----
#pragma unroll
    for (int off = 8; off > 0; off >>= 1) {
#pragma unroll
        for (int r = 0; r < 8; ++r) {
            const float ov = __shfl_xor(best[r], off, 32);
            const int   oi = __shfl_xor(bidx[r], off, 32);
            if (ov > best[r] || (ov == best[r] && oi < bidx[r])) {
                best[r] = ov; bidx[r] = oi;
            }
        }
    }

    // C/D layout: VGPR r holds M = r + 8*(lane/16); lane n==0 of each half
    // now holds the full argmax for its 8 rows.
    if (nloc == 0) {
#pragma unroll
        for (int r = 0; r < 8; ++r) {
            const int t = t0 + r + 8 * khalf;
            idx_out[(size_t)(b * TT + t) * ZZ + z] = bidx[r];
        }
    }
}

// ---------------------------------------------------------------------------
// Gather raw image rows: out[b,t,z,:] = image[b,z,idx[b,t,z],:]
// One block per (b,t,z); 64 threads x float4 = 256 floats.
// ---------------------------------------------------------------------------
__global__ __launch_bounds__(64) void gather_kernel(
    const float* __restrict__ image, const int* __restrict__ idx,
    float* __restrict__ out)
{
    const int r  = (int)blockIdx.x;      // (b*T + t)*Z + z
    const int z  = r & 15;
    const int bt = r >> 4;               // b*T + t
    const int b  = bt / TT;
    const int i  = idx[r];
    const float4* src = reinterpret_cast<const float4*>(
        image + ((size_t)(b * ZZ + z) * II + (size_t)i) * DD);
    float4* dst = reinterpret_cast<float4*>(out + (size_t)r * DD);
    dst[threadIdx.x] = src[threadIdx.x];
}

// ---------------------------------------------------------------------------
extern "C" void kernel_launch(void* const* d_in, const int* in_sizes, int n_in,
                              void* d_out, int out_size, void* d_ws, size_t ws_size,
                              hipStream_t stream)
{
    (void)in_sizes; (void)n_in; (void)out_size; (void)ws_size;
    const float* text  = (const float*)d_in[0];   // [B,T,D]   f32
    const float* image = (const float*)d_in[1];   // [B,Z,I,D] f32
    float* out = (float*)d_out;                   // [B,T,Z,D] f32

    int* idx = (int*)d_ws;                        // B*T*Z i32 (256 KB)

    // 1) WMMA cosine-similarity + argmax over I (norms fused into staging)
    argmax_wmma_kernel<<<BB * ZZ * (TT / 128), 256, 0, stream>>>(text, image, idx);

    // 2) gather raw image tokens
    gather_kernel<<<BB * TT * ZZ, 64, 0, stream>>>(image, idx, out);
}
